// CommNet_87067577024540
// MI455X (gfx1250) — matrix operand order repaired and checked by reading, not compiled, not run on hardware
//
#include <hip/hip_runtime.h>

// CommNet fused forward for MI455X (gfx1250, wave32, WMMA).
//   h = tanh(xs @ Wenc^T + benc)
//   3x: h = tanh(h @ (Wh - Wc/31)^T + tot @ (Wc/31)^T),  tot = sum_a h_a
//   out = h @ Wpol^T + bpol
// One block = 16 batch rows x all 32 agents (one wave per agent).
// All matmuls via v_wmma_f32_16x16x32_f16; h lives in LDS in WMMA A-operand
// layout so A loads are contiguous ds_load_b128 and the agent-sum is a
// linear elementwise reduction.

typedef __attribute__((ext_vector_type(16))) _Float16 v16h;
typedef __attribute__((ext_vector_type(8)))  float    v8f;
typedef __attribute__((ext_vector_type(4)))  float    v4f;

#define AGENTS 32
#define BATCH  16384
#define DIM    64
#define DACT   8

#if __has_builtin(__builtin_amdgcn_tanhf)
__device__ __forceinline__ float fast_tanh(float x) { return __builtin_amdgcn_tanhf(x); }
#else
__device__ __forceinline__ float fast_tanh(float x) {
  // tanh(x) = 1 - 2/(e^(2x)+1), using hardware v_exp_f32 / v_rcp_f32
  float e = __builtin_amdgcn_exp2f(2.8853900817779268f * x); // 2*log2(e)*x
  return 1.0f - 2.0f * __builtin_amdgcn_rcpf(e + 1.0f);
}
#endif

union H2 { unsigned u; _Float16 h[2]; };

// ---------------------------------------------------------------------------
// Prep kernel: fold weights, convert to f16, and emit the exact per-lane WMMA
// B-operand register image for each (slot m, N-tile t, K-half kb).
// B layout (16x16x32 f16): VGPR v of lane L holds B[K,N] with
//   K = kb*32 + (v>=4 ? 16 : 0) + (L>=16 ? 8 : 0) + 2(v&3) (+1 in hi half),  N = t*16 + (L%16).
// Slots: 0=Wenc, 1+2s=Wh'-s, 2+2s=Wc'-s, 7=Wpol (padded to 16 cols).
// Stored dword index == gid (layouts coincide). Total 64 KB in d_ws.
// ---------------------------------------------------------------------------
__global__ void prep_weights(const float* __restrict__ Wenc,
                             const float* __restrict__ Wh,
                             const float* __restrict__ Wc,
                             const float* __restrict__ Wpol,
                             unsigned* __restrict__ wbuf)
{
  int gid = blockIdx.x * blockDim.x + threadIdx.x;
  if (gid >= 8 * 2048) return;
  int v  = gid & 7;
  int L  = (gid >> 3) & 31;
  int kb = (gid >> 8) & 1;
  int t  = (gid >> 9) & 3;
  int m  = gid >> 11;
  int n  = t * 16 + (L & 15);
  int k  = kb * 32 + ((v >= 4) ? 16 : 0) + ((L >= 16) ? 8 : 0) + 2 * (v & 3);

  float w[2];
  #pragma unroll
  for (int q = 0; q < 2; ++q) {
    int kk = k + q;
    float val;
    if (m == 0) {
      val = Wenc[n * DIM + kk];
    } else if (m == 7) {
      val = (n < DACT) ? Wpol[n * DIM + kk] : 0.0f;
    } else {
      int s = (m - 1) >> 1;
      float wh = Wh[(s * DIM + n) * DIM + kk];
      float wc = Wc[(s * DIM + n) * DIM + kk];
      const float inv = 1.0f / 31.0f;   // norm = A-1
      val = (((m - 1) & 1) == 0) ? (wh - wc * inv) : (wc * inv);
    }
    w[q] = val;
  }
  H2 pk; pk.h[0] = (_Float16)w[0]; pk.h[1] = (_Float16)w[1];
  wbuf[gid] = pk.u;
}

// ---------------------------------------------------------------------------
// Main fused kernel.
// ---------------------------------------------------------------------------
#define WMMAF16(Aop, Bop, Cop) \
  __builtin_amdgcn_wmma_f32_16x16x32_f16(false, (Aop), false, (Bop), (short)0, (Cop), false, false)

__launch_bounds__(1024)
__global__ void commnet_fused(const float* __restrict__ xs,
                              const float* __restrict__ benc,
                              const float* __restrict__ bpol,
                              const unsigned* __restrict__ wbuf,
                              float* __restrict__ out)
{
  extern __shared__ char smem[];            // [0,65536): h per agent (A-op layout, 2KB each)
                                            // [65536,67584): tot buffer (same layout)
  const int lane = threadIdx.x;             // 0..31
  const int ag   = threadIdx.y;             // agent == wave id, 0..31
  const int li   = lane & 15;
  const int hi   = lane >> 4;
  const int b0   = blockIdx.x * 16;
  const int tid  = ag * 32 + lane;

  char*     hb     = smem + ag * 2048;
  char*     tb     = smem + 65536;
  unsigned* smem_u = (unsigned*)smem;
  unsigned* tb_u   = (unsigned*)tb;

  float btv[4];
  #pragma unroll
  for (int t = 0; t < 4; ++t) btv[t] = benc[t * 16 + li];
  float bp = (li < DACT) ? bpol[li] : 0.0f;

  // B operand (pre-swizzled in ws): contiguous 32B per lane -> global_load_b128 x2
  auto wop = [&](int m, int t, int kb) -> v16h {
    return *(const v16h*)((const char*)wbuf + (size_t)m * 8192 + (t * 2 + kb) * 1024 + lane * 32);
  };

  // Store one 16x16 C/D tile (acc layout: M = r + 8*hi, N = t*16 + li) into the
  // agent's LDS h-buffer at the position the NEXT layer's A-operand (K = N)
  // expects: kb = t/2, j = (t&1)*8 + (li&7), lane' = r + 8*hi + 16*(li>>3).
  auto store_tile = [&](int t, v8f acc, float bias) {
    #pragma unroll
    for (int r = 0; r < 8; ++r) {
      float x = fast_tanh(acc[r] + bias);
      int laneP   = r + 8 * hi + 16 * (li >> 3);
      int byteoff = (t >> 1) * 1024 + laneP * 32 + ((t & 1) * 8 + (li & 7)) * 2;
      *(_Float16*)(hb + byteoff) = (_Float16)x;
    }
  };

  // Elementwise cross-agent sum in f32, repacked to f16 (layouts identical).
  auto compute_tot = [&]() {
    if (tid < 512) {
      float s0 = 0.f, s1 = 0.f;
      #pragma unroll
      for (int aa = 0; aa < 32; ++aa) {
        H2 x; x.u = smem_u[aa * 512 + tid];
        s0 += (float)x.h[0];
        s1 += (float)x.h[1];
      }
      H2 r; r.h[0] = (_Float16)s0; r.h[1] = (_Float16)s1;
      tb_u[tid] = r.u;
    }
    __syncthreads();
  };

  // ---- encoder: build A operand straight from global xs (f32 -> f16) ----
  // A layout (16x32 f16): lane L, j=0..7 -> K = kb*32 + hi*8 + j ;
  //                               j=8..15 -> K = kb*32 + 16 + hi*8 + (j-8)
  v16h Ax0, Ax1;
  {
    const float* rowp = xs + ((size_t)ag * BATCH + b0 + li) * DIM;
    #pragma unroll
    for (int kb = 0; kb < 2; ++kb) {
      const float* p = rowp + kb * 32 + hi * 8;
      v4f f0 = *(const v4f*)(p + 0);
      v4f f1 = *(const v4f*)(p + 4);
      v4f g0 = *(const v4f*)(p + 16);
      v4f g1 = *(const v4f*)(p + 20);
      v16h Av;
      #pragma unroll
      for (int j = 0; j < 4; ++j) {
        Av[j]      = (_Float16)f0[j];
        Av[4 + j]  = (_Float16)f1[j];
        Av[8 + j]  = (_Float16)g0[j];
        Av[12 + j] = (_Float16)g1[j];
      }
      if (kb == 0) Ax0 = Av; else Ax1 = Av;
    }
  }

  #pragma unroll
  for (int t = 0; t < 4; ++t) {
    v8f acc = {};
    acc = WMMAF16(Ax0, wop(0, t, 0), acc);
    acc = WMMAF16(Ax1, wop(0, t, 1), acc);
    store_tile(t, acc, btv[t]);
  }
  __syncthreads();
  compute_tot();

  // ---- 3 communication steps ----
  #pragma unroll
  for (int s = 0; s < 3; ++s) {
    v16h Hh0 = *(const v16h*)(hb + lane * 32);          // own h, K=0..31
    v16h Hh1 = *(const v16h*)(hb + 1024 + lane * 32);   // own h, K=32..63
    v16h Tt0 = *(const v16h*)(tb + lane * 32);          // tot,   K=0..31
    v16h Tt1 = *(const v16h*)(tb + 1024 + lane * 32);   // tot,   K=32..63
    const int mh = 1 + 2 * s, mc = 2 + 2 * s;
    #pragma unroll
    for (int t = 0; t < 4; ++t) {
      v8f acc = {};
      acc = WMMAF16(Hh0, wop(mh, t, 0), acc);
      acc = WMMAF16(Hh1, wop(mh, t, 1), acc);
      acc = WMMAF16(Tt0, wop(mc, t, 0), acc);
      acc = WMMAF16(Tt1, wop(mc, t, 1), acc);
      store_tile(t, acc, 0.0f);
    }
    if (s < 2) {
      __syncthreads();
      compute_tot();
    }
  }

  // ---- policy head: single N-tile (cols 8..15 are zero-padded weights) ----
  v16h Hp0 = *(const v16h*)(hb + lane * 32);
  v16h Hp1 = *(const v16h*)(hb + 1024 + lane * 32);
  v8f acc = {};
  acc = WMMAF16(Hp0, wop(7, 0, 0), acc);
  acc = WMMAF16(Hp1, wop(7, 0, 1), acc);
  if (li < DACT) {
    #pragma unroll
    for (int r = 0; r < 8; ++r) {
      int row = r + 8 * hi;
      out[((size_t)ag * BATCH + b0 + row) * DACT + li] = acc[r] + bp;
    }
  }
}

// ---------------------------------------------------------------------------
extern "C" void kernel_launch(void* const* d_in, const int* in_sizes, int n_in,
                              void* d_out, int out_size, void* d_ws, size_t ws_size,
                              hipStream_t stream)
{
  (void)in_sizes; (void)n_in; (void)out_size; (void)ws_size;
  const float* xs   = (const float*)d_in[0];
  const float* Wenc = (const float*)d_in[1];
  const float* benc = (const float*)d_in[2];
  const float* Wh   = (const float*)d_in[3];
  const float* Wc   = (const float*)d_in[4];
  const float* Wpol = (const float*)d_in[5];
  const float* bpol = (const float*)d_in[6];
  float*    out  = (float*)d_out;
  unsigned* wbuf = (unsigned*)d_ws;   // 64 KB of pre-swizzled f16 weights

  prep_weights<<<64, 256, 0, stream>>>(Wenc, Wh, Wc, Wpol, wbuf);

  const int smem_bytes = 65536 + 2048;  // 32 agents x 2KB h + 2KB tot
  (void)hipFuncSetAttribute(reinterpret_cast<const void*>(commnet_fused),
                            hipFuncAttributeMaxDynamicSharedMemorySize, smem_bytes);
  dim3 blk(32, 32, 1);                  // 32 lanes x 32 waves (one wave per agent)
  commnet_fused<<<BATCH / 16, blk, smem_bytes, stream>>>(xs, benc, bpol, wbuf, out);
}
